// BertClassifier_74749610820199
// MI455X (gfx1250) — compile-verified, matrix operand order checked by reading, not run on hardware
//
#include <hip/hip_runtime.h>
#include <hip/hip_bf16.h>
#include <math.h>

// ---- problem constants ----
#define L      252
#define HID    768
#define MLP    770
#define NLAB   36
#define B      2
#define KPAD   800        // MLP padded to 25 chunks of 32
#define NPAD   48         // labels padded to 3 tiles of 16
#define MROWS  512        // B*L = 504 padded to 32 m-tiles
#define NROWS  1600       // [W1i | pad | W1j | pad] rows
#define PPOS   (L*L)      // 63504 positions per (b,label)

typedef _Float16 half8 __attribute__((ext_vector_type(8)));
typedef _Float16 v16h  __attribute__((ext_vector_type(16)));
typedef float    v8f   __attribute__((ext_vector_type(8)));

#define SHUF16(lo, hi) __builtin_shufflevector(lo, hi, 0,1,2,3,4,5,6,7,8,9,10,11,12,13,14,15)

// ---------------- prep: f32 -> f16 staging (zero padded) ----------------
__global__ __launch_bounds__(256) void prep_hf16(const float* __restrict__ hidden,
                                                 _Float16* __restrict__ hF) {
  int idx = blockIdx.x * 256 + threadIdx.x;
  if (idx >= MROWS * HID) return;
  int m = idx / HID, d = idx % HID;
  _Float16 v = (_Float16)0.f;
  if (m < B * L) {
    int b = m / L, i = m % L;
    v = (_Float16)hidden[((size_t)b * (L + 1) + (i + 1)) * HID + d];
  }
  hF[idx] = v;
}

__global__ __launch_bounds__(256) void prep_w1f16(const float* __restrict__ W1,
                                                  _Float16* __restrict__ W1h) {
  int idx = blockIdx.x * 256 + threadIdx.x;
  if (idx >= NROWS * HID) return;
  int n = idx / HID, d = idx % HID;
  float v = 0.f;
  if (n < MLP)                       v = W1[(size_t)n * (2 * HID + 1) + d];          // W1i
  else if (n >= KPAD && n < KPAD+MLP) v = W1[(size_t)(n - KPAD) * (2 * HID + 1) + HID + d]; // W1j
  W1h[idx] = (_Float16)v;
}

__global__ __launch_bounds__(256) void prep_w2f16(const float* __restrict__ W2,
                                                  _Float16* __restrict__ W2h) {
  int idx = blockIdx.x * 256 + threadIdx.x;
  if (idx >= NPAD * KPAD) return;
  int k = idx / KPAD, h = idx % KPAD;
  float v = (k < NLAB && h < MLP) ? W2[(size_t)k * MLP + h] : 0.f;
  W2h[idx] = (_Float16)v;
}

__global__ __launch_bounds__(256) void prep_w1c_padAj(const float* __restrict__ W1,
                                                      _Float16* __restrict__ w1c,
                                                      _Float16* __restrict__ Aj) {
  int idx = blockIdx.x * 256 + threadIdx.x;
  if (idx < KPAD)
    w1c[idx] = (idx < MLP) ? (_Float16)W1[(size_t)idx * (2 * HID + 1) + 2 * HID] : (_Float16)0.f;
  // zero Aj pad rows j = 252..255 for both batches (read by stage 2's last j-tile)
  int pidx = idx - KPAD;
  if (pidx >= 0 && pidx < B * 4 * KPAD) {
    int b = pidx / (4 * KPAD);
    int r = L + (pidx / KPAD) % 4;
    int c = pidx % KPAD;
    Aj[((size_t)(b * 256 + r)) * KPAD + c] = (_Float16)0.f;
  }
}

// ---------------- stage 1: Ai/Aj via WMMA --------------------------------
// C(512 x 1600) = hF(512 x 768) * W1h^T ; cols [0,800) -> Ai (+b1), [800,1600) -> Aj
__global__ __launch_bounds__(128) void stage1_wmma(const _Float16* __restrict__ hF,
                                                   const _Float16* __restrict__ W1h,
                                                   const float* __restrict__ b1,
                                                   _Float16* __restrict__ Ai,
                                                   _Float16* __restrict__ Aj) {
  const int lane = threadIdx.x & 31;
  const int wid  = threadIdx.x >> 5;
  const int nt   = blockIdx.x * 4 + wid;   // 0..99
  const int mt   = blockIdx.y;             // 0..31
  const int jl   = lane & 15;
  const int kg   = lane >> 4;

  const _Float16* arow = hF  + (size_t)(mt * 16 + jl) * HID;
  const _Float16* brow = W1h + (size_t)(nt * 16 + jl) * HID;

  v8f acc = {};
#pragma unroll 4
  for (int kc = 0; kc < 24; ++kc) {
    int r0 = kc * 32 + kg * 8;
    int r1 = r0 + 16;
    v16h A = SHUF16(*(const half8*)(arow + r0), *(const half8*)(arow + r1));
    v16h Bm = SHUF16(*(const half8*)(brow + r0), *(const half8*)(brow + r1));
    acc = __builtin_amdgcn_wmma_f32_16x16x32_f16(false, A, false, Bm, (short)0, acc, false, false);
  }

  const int n = nt * 16 + jl;
#pragma unroll
  for (int v = 0; v < 8; ++v) {
    int m = mt * 16 + ((lane < 16) ? v : v + 8);
    if (m >= B * L) continue;
    int b = m / L, ii = m % L;
    if (n < KPAD) {
      float val = acc[v] + ((n < MLP) ? b1[n] : 0.f);
      Ai[((size_t)(b * 256 + ii)) * KPAD + n] = (n < MLP) ? (_Float16)val : (_Float16)0.f;
    } else {
      int c = n - KPAD;
      Aj[((size_t)(b * 256 + ii)) * KPAD + c] = (c < MLP) ? (_Float16)acc[v] : (_Float16)0.f;
    }
  }
}

// ---------------- stage 2: the big (127008 x 770) x (770 x 36) GEMM ------
// one wave per (b, i, 16-wide j tile); A fragment synthesized with packed f16.
__global__ __launch_bounds__(128) void stage2_wmma(const _Float16* __restrict__ Ai,
                                                   const _Float16* __restrict__ Aj,
                                                   const _Float16* __restrict__ w1c,
                                                   const _Float16* __restrict__ W2h,
                                                   const float* __restrict__ b2,
                                                   const int* __restrict__ spans,
                                                   const int* __restrict__ smask,
                                                   float* __restrict__ out) {
  const int lane = threadIdx.x & 31;
  const int wid  = threadIdx.x >> 5;
  const int jt   = blockIdx.x * 4 + wid;   // 0..15
  const int i    = blockIdx.y;             // 0..251
  const int b    = blockIdx.z;             // 0..1
  const int j0   = jt * 16;
  const int jl   = lane & 15;
  const int kg   = lane >> 4;
  const int jrow = j0 + jl;                // A-matrix row owned by this lane

  const int s = spans[b * 2 + 0];
  const int e = spans[b * 2 + 1];
  float indf = ((i == s) && (jrow == e)) ? 2.0f
             : ((s <= i) && (i <= jrow) && (jrow <= e)) ? 1.0f : 0.0f;
  _Float16 ih = (_Float16)indf;
  half8 indv = {ih, ih, ih, ih, ih, ih, ih, ih};
  half8 z8 = {};

  const _Float16* aiRow = Ai + (size_t)(b * 256 + i)    * KPAD;
  const _Float16* ajRow = Aj + (size_t)(b * 256 + jrow) * KPAD;
  const _Float16* w0row = W2h + (size_t)(0 * 16 + jl) * KPAD;
  const _Float16* w1row = W2h + (size_t)(1 * 16 + jl) * KPAD;
  const _Float16* w2row = W2h + (size_t)(2 * 16 + jl) * KPAD;

  v8f acc0 = {}, acc1 = {}, acc2 = {};
#pragma unroll 5
  for (int kc = 0; kc < 25; ++kc) {
    int r0 = kc * 32 + kg * 8;
    int r1 = r0 + 16;
    half8 h0 = *(const half8*)(aiRow + r0) + *(const half8*)(ajRow + r0)
             + *(const half8*)(w1c + r0) * indv;
    half8 h1 = *(const half8*)(aiRow + r1) + *(const half8*)(ajRow + r1)
             + *(const half8*)(w1c + r1) * indv;
    h0 = __builtin_elementwise_max(h0, z8);   // relu -> v_pk_max
    h1 = __builtin_elementwise_max(h1, z8);
    v16h A = SHUF16(h0, h1);

    v16h B0 = SHUF16(*(const half8*)(w0row + r0), *(const half8*)(w0row + r1));
    acc0 = __builtin_amdgcn_wmma_f32_16x16x32_f16(false, A, false, B0, (short)0, acc0, false, false);
    v16h B1 = SHUF16(*(const half8*)(w1row + r0), *(const half8*)(w1row + r1));
    acc1 = __builtin_amdgcn_wmma_f32_16x16x32_f16(false, A, false, B1, (short)0, acc1, false, false);
    v16h B2 = SHUF16(*(const half8*)(w2row + r0), *(const half8*)(w2row + r1));
    acc2 = __builtin_amdgcn_wmma_f32_16x16x32_f16(false, A, false, B2, (short)0, acc2, false, false);
  }

  // store transposed pre-softmax: out[b][k][i*252 + j], mask + b2 applied
  const int* mrow = smask + (size_t)i * L;
#pragma unroll
  for (int nt = 0; nt < 3; ++nt) {
    int k = nt * 16 + jl;                   // D-matrix N = lane%16
    if (k >= NLAB) continue;
    float bb = b2[k];
    float* orow = out + ((size_t)(b * NLAB + k)) * PPOS + (size_t)i * L;
    const v8f& a = (nt == 0) ? acc0 : (nt == 1) ? acc1 : acc2;
#pragma unroll
    for (int v = 0; v < 8; ++v) {
      int jj = j0 + ((lane < 16) ? v : v + 8);   // D-matrix M row
      if (jj < L) {
        float val = a[v] + bb;
        if (mrow[jj] < 1) val = 0.f;
        orow[jj] = val;
      }
    }
  }
}

// ---------------- stage 3: in-place log-softmax over the 63504 axis ------
__global__ __launch_bounds__(256) void logsoftmax_rows(float* __restrict__ out) {
  const int row = blockIdx.x;               // 0..71 = (b,k)
  float* p = out + (size_t)row * PPOS;
  __shared__ float red[256];
  const int t = threadIdx.x;

  float m = -3.4e38f;
  for (int x = t; x < PPOS; x += 256) m = fmaxf(m, p[x]);
  red[t] = m; __syncthreads();
  for (int o = 128; o > 0; o >>= 1) { if (t < o) red[t] = fmaxf(red[t], red[t + o]); __syncthreads(); }
  m = red[0]; __syncthreads();

  float ssum = 0.f;
  for (int x = t; x < PPOS; x += 256) ssum += expf(p[x] - m);
  red[t] = ssum; __syncthreads();
  for (int o = 128; o > 0; o >>= 1) { if (t < o) red[t] += red[t + o]; __syncthreads(); }
  float lse = m + logf(red[0]);

  for (int x = t; x < PPOS; x += 256) p[x] -= lse;
}

// ---------------- host launcher ------------------------------------------
extern "C" void kernel_launch(void* const* d_in, const int* in_sizes, int n_in,
                              void* d_out, int out_size, void* d_ws, size_t ws_size,
                              hipStream_t stream) {
  (void)in_sizes; (void)n_in; (void)out_size; (void)ws_size;
  const float* hidden = (const float*)d_in[0];
  const int*   spans  = (const int*)  d_in[1];
  const int*   smask  = (const int*)  d_in[2];
  const float* W1     = (const float*)d_in[3];
  const float* b1     = (const float*)d_in[4];
  const float* W2     = (const float*)d_in[5];
  const float* b2     = (const float*)d_in[6];
  float* out = (float*)d_out;

  size_t off = 0;
  auto carve = [&](size_t bytes) -> void* {
    void* p = (char*)d_ws + off;
    off += (bytes + 255) & ~(size_t)255;
    return p;
  };
  _Float16* hF  = (_Float16*)carve((size_t)MROWS * HID * 2);
  _Float16* W1h = (_Float16*)carve((size_t)NROWS * HID * 2);
  _Float16* W2h = (_Float16*)carve((size_t)NPAD * KPAD * 2);
  _Float16* w1c = (_Float16*)carve((size_t)KPAD * 2);
  _Float16* Ai  = (_Float16*)carve((size_t)B * 256 * KPAD * 2);
  _Float16* Aj  = (_Float16*)carve((size_t)B * 256 * KPAD * 2);

  prep_hf16 <<<(MROWS * HID + 255) / 256, 256, 0, stream>>>(hidden, hF);
  prep_w1f16<<<(NROWS * HID + 255) / 256, 256, 0, stream>>>(W1, W1h);
  prep_w2f16<<<(NPAD * KPAD + 255) / 256, 256, 0, stream>>>(W2, W2h);
  prep_w1c_padAj<<<((KPAD + B * 4 * KPAD) + 255) / 256, 256, 0, stream>>>(W1, w1c, Aj);

  stage1_wmma<<<dim3(25, 32), 128, 0, stream>>>(hF, W1h, b1, Ai, Aj);
  stage2_wmma<<<dim3(4, L, B), 128, 0, stream>>>(Ai, Aj, w1c, W2h, b2, spans, smask, out);
  logsoftmax_rows<<<B * NLAB, 256, 0, stream>>>(out);
}
